// DCNSepHead_13692355740057
// MI455X (gfx1250) — matrix operand order repaired and checked by reading, not compiled, not run on hardware
//
#include <hip/hip_runtime.h>

// ---------------------------------------------------------------------------
// CDNA5 (gfx1250) implementation of DCN + SepHead reference.
// Dominant work (deform conv implicit GEMM, ~155 GFLOP) runs on
// v_wmma_f32_16x16x32_bf16 with fp32 accumulation.
// Round 3: vectorized (u32x4) MODE-0 A staging to remove the serialized
// per-element global_load_u16/wait pattern; unrolled staging loops.
// ---------------------------------------------------------------------------

#define HW        16384      // 128*128 pixels
#define IMW       128
#define PIX_TILE  128        // pixels per workgroup tile
#define KCH       32         // K-chunk per WMMA (bf16 K=32)
#define A_STRIDE  40         // bf16 elems per A_lds row (80B, 16B aligned)
#define B_STRIDE  36         // bf16 elems per B_lds row (72B, 8B aligned)

typedef __bf16        v16bf  __attribute__((ext_vector_type(16)));
typedef float         v8f    __attribute__((ext_vector_type(8)));
typedef unsigned int  u32x4  __attribute__((ext_vector_type(4)));
typedef unsigned int  u32x2  __attribute__((ext_vector_type(2)));

union AFrag { u32x4 q[2]; v16bf v; };
union BFrag { u32x2 d[4]; v16bf v; };

__device__ __forceinline__ unsigned short f2bf(float f) {
  // round-to-nearest-even fp32 -> bf16
  unsigned int u = __float_as_uint(f);
  unsigned int r = u + 0x7fffu + ((u >> 16) & 1u);
  return (unsigned short)(r >> 16);
}

// ---------------------------------------------------------------------------
// fp32 -> bf16 converter
// ---------------------------------------------------------------------------
__global__ void cvt_bf16_kernel(const float* __restrict__ in,
                                unsigned short* __restrict__ out, int n) {
  int i = blockIdx.x * 256 + threadIdx.x;
  if (i < n) out[i] = f2bf(in[i]);
}

// ---------------------------------------------------------------------------
// Staging helpers
// ---------------------------------------------------------------------------
template <int NSUB>
__device__ __forceinline__ void stage_B(unsigned short* __restrict__ Bl,
                                        int tid, int oc0, int kb,
                                        const unsigned short* __restrict__ Wbf,
                                        int Nout, int Kdim) {
#pragma unroll
  for (int f = tid; f < NSUB * 16 * 8; f += 256) {   // one u32x2 (4 bf16) per f
    int n  = f >> 3;
    int k4 = (f & 7) << 2;
    int gn = oc0 + n;
    u32x2 v = {0u, 0u};
    if (gn < Nout)
      v = *(const u32x2*)(Wbf + (size_t)gn * Kdim + kb + k4);
    *(u32x2*)(&Bl[n * B_STRIDE + k4]) = v;
  }
}

template <int MODE>
__device__ __forceinline__ void stage_A(unsigned short* __restrict__ Al,
                                        int tid, int pix0, int kb,
                                        const unsigned short* __restrict__ Abf,
                                        const float* __restrict__ xf32,
                                        const float* __restrict__ offs) {
  if constexpr (MODE == 2) {
    // deformable bilinear gather (inherently scalar)
    for (int f = tid; f < PIX_TILE * KCH; f += 256) {
      int p  = f >> 5;
      int kk = f & 31;
      int kg = kb + kk;                 // global K = (g*128+c)*9 + tap
      int g   = kg / 1152;
      int r0  = kg - g * 1152;
      int c   = r0 / 9;
      int tap = r0 - c * 9;
      int P  = pix0 + p;
      int h  = P >> 7, wc = P & 127;
      int ch = (g * 9 + tap) * 2;
      float dy = offs[(size_t)ch * HW + P];
      float dx = offs[(size_t)(ch + 1) * HW + P];
      float sy = (float)(h + tap / 3 - 1) + dy;
      float sx = (float)(wc + tap % 3 - 1) + dx;
      float y0f = floorf(sy), x0f = floorf(sx);
      float wy = sy - y0f, wx = sx - x0f;
      int y0 = (int)y0f, x0 = (int)x0f;
      const float* plane = xf32 + (size_t)(g * 128 + c) * HW;
      float v = 0.0f;
#pragma unroll
      for (int dyi = 0; dyi < 2; ++dyi)
#pragma unroll
        for (int dxi = 0; dxi < 2; ++dxi) {
          int yi = y0 + dyi, xi = x0 + dxi;
          float wgt = (dyi ? wy : 1.0f - wy) * (dxi ? wx : 1.0f - wx);
          if (yi >= 0 && yi < IMW && xi >= 0 && xi < IMW)
            v += wgt * plane[yi * IMW + xi];
        }
      Al[p * A_STRIDE + kk] = f2bf(v);
    }
  } else if constexpr (MODE == 0) {
    // contiguous row copy: one u32x4 = 8 bf16 pixels per load, coalesced
#pragma unroll
    for (int f = tid; f < PIX_TILE * KCH / 8; f += 256) {  // 512 items, 2 iters
      int p8 = (f & 15) << 3;           // pixel group 0..120 step 8
      int k  = f >> 4;                  // 0..31
      int kg = kb + k;
      u32x4 v = *(const u32x4*)(Abf + (size_t)kg * HW + pix0 + p8);
      unsigned short* dst = &Al[p8 * A_STRIDE + k];
#pragma unroll
      for (int e = 0; e < 4; ++e) {
        unsigned int w = v[e];
        dst[(2 * e + 0) * A_STRIDE] = (unsigned short)(w & 0xffffu);
        dst[(2 * e + 1) * A_STRIDE] = (unsigned short)(w >> 16);
      }
    }
  } else {
    // MODE 1: im2col 3x3, pad 1 (column shifts prevent vector loads)
    for (int f = tid; f < PIX_TILE * KCH; f += 256) {
      int p  = f & 127;                 // lanes -> consecutive pixels
      int k  = f >> 7;
      int kg = kb + k;
      int P  = pix0 + p;
      int c = kg / 9, tap = kg - c * 9;
      int h = P >> 7, wc = P & 127;
      int yy = h + tap / 3 - 1, xx = wc + tap % 3 - 1;
      unsigned short v = (yy >= 0 && yy < IMW && xx >= 0 && xx < IMW)
              ? Abf[(size_t)c * HW + yy * IMW + xx] : (unsigned short)0;
      Al[p * A_STRIDE + k] = v;
    }
  }
}

// ---------------------------------------------------------------------------
// Generic WMMA conv-as-GEMM kernel.
//   out[n][p] = act( BN( sum_k W[n][k] * A(k,p) + bias[n] ) )
// MODE 0: A(k,p) = Abf[k*HW + p]                       (1x1 conv)
// MODE 1: A(k,p) = im2col 3x3 of Abf (k = c*9+tap)     (3x3 conv, pad 1)
// MODE 2: A(k,p) = bilinear deform gather from xf32    (k = (g*128+c)*9+tap)
// Tile: 128 pixels x (NSUB*16) out-channels, 8 waves, wave = 16-pixel stripe.
// Double-buffered LDS: compute chunk i while staging chunk i+1.
// ---------------------------------------------------------------------------
template <int MODE, int NSUB>
__global__ __launch_bounds__(256) void wmma_conv(
    const unsigned short* __restrict__ Abf,   // bf16 activations [K][HW] (MODE 0/1)
    const float*          __restrict__ xf32,  // fp32 gather source [512][HW] (MODE 2)
    const float*          __restrict__ offs,  // offsets [72][HW] (MODE 2)
    const unsigned short* __restrict__ Wbf,   // bf16 weights [N][K]
    const float* __restrict__ bias,           // [N] or nullptr
    const float* __restrict__ gamma,          // BN params or nullptr
    const float* __restrict__ beta,
    const float* __restrict__ mean,
    const float* __restrict__ var,
    float* __restrict__ out,                  // [N][HW]
    int Nout, int Kdim, int relu)
{
  __shared__ unsigned short Alds[2][PIX_TILE * A_STRIDE];
  __shared__ unsigned short Blds[2][NSUB * 16 * B_STRIDE];

  const int tid  = threadIdx.x;
  const int lane = tid & 31;
  const int wave = tid >> 5;
  const int pix0 = blockIdx.x * PIX_TILE;
  const int oc0  = blockIdx.y * (NSUB * 16);
  const int m    = lane & 15;   // A: row M ; B/D: column N
  const int hi   = lane >> 4;   // half-wave select

  v8f acc[NSUB] = {};

  // prologue: stage chunk 0 into buffer 0
  stage_A<MODE>(Alds[0], tid, pix0, 0, Abf, xf32, offs);
  stage_B<NSUB>(Blds[0], tid, oc0, 0, Wbf, Nout, Kdim);

  const int nk = Kdim / KCH;
  for (int i = 0; i < nk; ++i) {
    __syncthreads();                    // staging of buffer (i&1) complete
    const int cur = i & 1;
    const unsigned short* Al = Alds[cur];
    const unsigned short* Bl = Blds[cur];

    // A fragment: 16x32 bf16, lane(M=m,hi): elems 0..7 -> K=hi*8+0..7,
    //                                        elems 8..15 -> K=16+hi*8+0..7
    AFrag ua;
    const unsigned short* arow = &Al[(wave * 16 + m) * A_STRIDE + hi * 8];
    ua.q[0] = *(const u32x4*)(arow);    // 80B rows -> 16B aligned
    ua.q[1] = *(const u32x4*)(arow + 16);

    // B fragments preloaded in groups of 4; WMMAs issue back-to-back, each
    // group's ds loads overlap the previous group's XDL execution.
#pragma unroll
    for (int g = 0; g < NSUB; g += 4) {
      BFrag ub[4];
#pragma unroll
      for (int s = 0; s < 4; ++s) {
        // B 32x16 bf16: lane(N=m,hi): elems 0..7 -> K = hi*8 + e, elems 8..15 -> K = 16 + hi*8 + (e-8)
        const unsigned short* brow = &Bl[((g + s) * 16 + m) * B_STRIDE + hi * 8];
        ub[s].d[0] = *(const u32x2*)(brow + 0);
        ub[s].d[1] = *(const u32x2*)(brow + 4);
        ub[s].d[2] = *(const u32x2*)(brow + 16);
        ub[s].d[3] = *(const u32x2*)(brow + 20);
      }
#pragma unroll
      for (int s = 0; s < 4; ++s)
        acc[g + s] = __builtin_amdgcn_wmma_f32_16x16x32_bf16(
            false, ua.v, false, ub[s].v, (short)0, acc[g + s], false, false);
    }

    // stage next chunk into the other buffer (overlaps WMMA across waves)
    if (i + 1 < nk) {
      stage_A<MODE>(Alds[cur ^ 1], tid, pix0, (i + 1) * KCH, Abf, xf32, offs);
      stage_B<NSUB>(Blds[cur ^ 1], tid, oc0, (i + 1) * KCH, Wbf, Nout, Kdim);
    }
  }

  // ---- epilogue: bias + BN + ReLU, D lane -> (N = m, M = r + hi*8) ----
#pragma unroll
  for (int s = 0; s < NSUB; ++s) {
    int n = oc0 + s * 16 + m;
    if (n >= Nout) continue;
    float b = bias ? bias[n] : 0.0f;
    float sc = 1.0f, sh = 0.0f;
    if (gamma) {
      sc = gamma[n] * rsqrtf(var[n] + 1e-5f);
      sh = beta[n] - mean[n] * sc;
    }
#pragma unroll
    for (int r = 0; r < 8; ++r) {
      float v = (acc[s][r] + b) * sc + sh;
      if (relu) v = fmaxf(v, 0.0f);
      out[(size_t)n * HW + pix0 + wave * 16 + hi * 8 + r] = v;
    }
  }
}

// ---------------------------------------------------------------------------
// Tiny final convs (negligible FLOPs) — plain fp32.
// ---------------------------------------------------------------------------
__global__ void conv3x3_64_1(const float* __restrict__ t,
                             const float* __restrict__ w,
                             const float* __restrict__ b,
                             float* __restrict__ out) {
  int P = blockIdx.x * 256 + threadIdx.x;
  int h = P >> 7, wc = P & 127;
  float acc = b[0];
  for (int c = 0; c < 64; ++c)
#pragma unroll
    for (int tap = 0; tap < 9; ++tap) {
      int yy = h + tap / 3 - 1, xx = wc + tap % 3 - 1;
      if (yy >= 0 && yy < IMW && xx >= 0 && xx < IMW)
        acc += w[c * 9 + tap] * t[(size_t)c * HW + yy * IMW + xx];
    }
  out[P] = acc;
}

__global__ void conv1x1_small(const float* __restrict__ t,
                              const float* __restrict__ w,
                              const float* __restrict__ b,
                              float* __restrict__ out, int ncls) {
  int P = blockIdx.x * 256 + threadIdx.x;
  for (int o = 0; o < ncls; ++o) {
    float acc = b[o];
    for (int c = 0; c < 64; ++c) acc += w[o * 64 + c] * t[(size_t)c * HW + P];
    out[(size_t)o * HW + P] = acc;
  }
}

// ---------------------------------------------------------------------------
// Host orchestration
// ---------------------------------------------------------------------------
extern "C" void kernel_launch(void* const* d_in, const int* in_sizes, int n_in,
                              void* d_out, int out_size, void* d_ws, size_t ws_size,
                              hipStream_t stream) {
  (void)in_sizes; (void)n_in; (void)out_size; (void)ws_size;

  const float* x = (const float*)d_in[0];
  const float* offw[2] = {(const float*)d_in[1], (const float*)d_in[4]};
  const float* offb[2] = {(const float*)d_in[2], (const float*)d_in[5]};
  const float* dcnw[2] = {(const float*)d_in[3], (const float*)d_in[6]};
  const float* cw1 = (const float*)d_in[7];
  const float* cb1 = (const float*)d_in[8];
  const float* cg  = (const float*)d_in[9];
  const float* cbe = (const float*)d_in[10];
  const float* cme = (const float*)d_in[11];
  const float* cva = (const float*)d_in[12];
  const float* cw2 = (const float*)d_in[13];
  const float* cb2 = (const float*)d_in[14];
  // heads: reg, height, dim, rot, vel — 8 tensors each starting at 15
  const int head_base = 15;
  const int head_choff[5] = {1, 3, 4, 7, 9};
  const int head_ncls[5]  = {2, 1, 3, 2, 2};

  float* out = (float*)d_out;

  // ---- workspace carve-up (256B aligned) ----
  char* wp = (char*)d_ws;
  auto carve = [&](size_t bytes) -> char* {
    char* r = wp;
    wp += (bytes + 255) & ~(size_t)255;
    return r;
  };
  unsigned short* xbf    = (unsigned short*)carve((size_t)512 * HW * 2);
  unsigned short* featbf = (unsigned short*)carve((size_t)512 * HW * 2);
  unsigned short* wdcn0  = (unsigned short*)carve((size_t)512 * 4608 * 2);
  unsigned short* wdcn1  = (unsigned short*)carve((size_t)512 * 4608 * 2);
  unsigned short* wcls1  = (unsigned short*)carve((size_t)64 * 4608 * 2);
  unsigned short* woff0  = (unsigned short*)carve((size_t)72 * 512 * 2);
  unsigned short* woff1  = (unsigned short*)carve((size_t)72 * 512 * 2);
  unsigned short* whead  = (unsigned short*)carve((size_t)5 * 64 * 512 * 2);
  float* off0  = (float*)carve((size_t)72 * HW * 4);
  float* off1  = (float*)carve((size_t)72 * HW * 4);
  float* cfeat = (float*)carve((size_t)512 * HW * 4);
  float* rfeat = (float*)carve((size_t)512 * HW * 4);
  float* tbuf  = (float*)carve((size_t)64 * HW * 4);

  auto cvt = [&](const float* src, unsigned short* dst, int n) {
    cvt_bf16_kernel<<<(n + 255) / 256, 256, 0, stream>>>(src, dst, n);
  };

  // ---- stage 0: bf16 conversions of x and all GEMM weights ----
  cvt(x, xbf, 512 * HW);
  cvt(dcnw[0], wdcn0, 512 * 4608);
  cvt(dcnw[1], wdcn1, 512 * 4608);
  cvt(cw1, wcls1, 64 * 4608);
  cvt(offw[0], woff0, 72 * 512);
  cvt(offw[1], woff1, 72 * 512);
  for (int i = 0; i < 5; ++i)
    cvt((const float*)d_in[head_base + i * 8 + 0], whead + (size_t)i * 64 * 512, 64 * 512);

  // ---- stage 1: offset convs (1x1, 512 -> 72), both adapts ----
  wmma_conv<0, 8><<<dim3(HW / PIX_TILE, 1), 256, 0, stream>>>(
      xbf, nullptr, nullptr, woff0, offb[0],
      nullptr, nullptr, nullptr, nullptr, off0, 72, 512, 0);
  wmma_conv<0, 8><<<dim3(HW / PIX_TILE, 1), 256, 0, stream>>>(
      xbf, nullptr, nullptr, woff1, offb[1],
      nullptr, nullptr, nullptr, nullptr, off1, 72, 512, 0);

  // ---- stage 2: deformable conv GEMMs (K=4608, N=512) + ReLU ----
  wmma_conv<2, 8><<<dim3(HW / PIX_TILE, 4), 256, 0, stream>>>(
      nullptr, x, off0, wdcn0, nullptr,
      nullptr, nullptr, nullptr, nullptr, cfeat, 512, 4608, 1);
  wmma_conv<2, 8><<<dim3(HW / PIX_TILE, 4), 256, 0, stream>>>(
      nullptr, x, off1, wdcn1, nullptr,
      nullptr, nullptr, nullptr, nullptr, rfeat, 512, 4608, 1);

  // ---- stage 3: cls head: 3x3 conv 512->64 + BN + ReLU, then 3x3 64->1 ----
  cvt(cfeat, featbf, 512 * HW);
  wmma_conv<1, 4><<<dim3(HW / PIX_TILE, 1), 256, 0, stream>>>(
      featbf, nullptr, nullptr, wcls1, cb1,
      cg, cbe, cme, cva, tbuf, 64, 4608, 1);
  conv3x3_64_1<<<HW / 256, 256, 0, stream>>>(tbuf, cw2, cb2, out /*channel 0*/);

  // ---- stage 4: SepHead task heads on reg_feat ----
  cvt(rfeat, featbf, 512 * HW);
  for (int i = 0; i < 5; ++i) {
    const float* b1 = (const float*)d_in[head_base + i * 8 + 1];
    const float* g  = (const float*)d_in[head_base + i * 8 + 2];
    const float* be = (const float*)d_in[head_base + i * 8 + 3];
    const float* me = (const float*)d_in[head_base + i * 8 + 4];
    const float* va = (const float*)d_in[head_base + i * 8 + 5];
    const float* w2 = (const float*)d_in[head_base + i * 8 + 6];
    const float* b2 = (const float*)d_in[head_base + i * 8 + 7];
    wmma_conv<0, 4><<<dim3(HW / PIX_TILE, 1), 256, 0, stream>>>(
        featbf, nullptr, nullptr, whead + (size_t)i * 64 * 512, b1,
        g, be, me, va, tbuf, 64, 512, 1);
    conv1x1_small<<<HW / 256, 256, 0, stream>>>(
        tbuf, w2, b2, out + (size_t)head_choff[i] * HW, head_ncls[i]);
  }
}
